// CIFModule_22789096473054
// MI455X (gfx1250) — matrix operand order repaired and checked by reading, not compiled, not run on hardware
//
#include <hip/hip_runtime.h>

// ---------------------------------------------------------------------------
// CIF module for MI455X (gfx1250, wave32).
// GEMMs run on v_wmma_f32_16x16x32_bf16 with fp32 accumulate, 8 N-tiles per
// wave (8 WMMAs per A-fragment fetch). The three output projections are fused
// into one K=512 GEMM over a concatenated bf16 activation buffer, removing
// two read-modify-write passes (~160MB) over the 39MB output.
// ---------------------------------------------------------------------------

#define B_      64
#define T_BI    3000
#define D_BI    128
#define T_SW    375
#define D_SW    192
#define T_S1    188
#define N_FIRES 300
#define D_MODEL 512
#define M_ROWS  (B_ * N_FIRES)            // 19200, multiple of 16
#define EMB_SZ  (M_ROWS * D_MODEL)        // 9,830,400
#define ALPHA_SZ (B_ * T_BI)              // 192,000
#define K_CAT   512                       // 128 + 192 + 192
#define SCALE_F 4.0f
#define EPS_LN  1e-5f

typedef __attribute__((ext_vector_type(16))) __bf16 v16bf;
typedef __attribute__((ext_vector_type(8)))  float  v8f;

static __device__ __forceinline__ unsigned short f2bf(float f) {
  union { float f; unsigned u; } v; v.f = f;
  unsigned u = v.u;
  u += 0x7FFFu + ((u >> 16) & 1u);        // round to nearest even
  return (unsigned short)(u >> 16);
}

static __device__ __forceinline__ int lower_bound(const float* c, int n, float v) {
  int lo = 0, hi = n;
  while (lo < hi) { int m = (lo + hi) >> 1; if (c[m] < v) lo = m + 1; else hi = m; }
  return lo;
}

// ---------------------------------------------------------------------------
// 1) Weight predictor: depthwise scale -> LayerNorm(128) -> dot -> sigmoid*4.
// ---------------------------------------------------------------------------
__global__ void k_alpha(const float* __restrict__ fs,
                        const float* __restrict__ conv_w,
                        const float* __restrict__ ln_g,
                        const float* __restrict__ ln_b,
                        const float* __restrict__ wp_w,
                        const float* __restrict__ wp_b,
                        float* __restrict__ alpha_out) {
  int wave = blockIdx.x * (blockDim.x >> 5) + (threadIdx.x >> 5);  // row id
  int lane = threadIdx.x & 31;
  const float4 xv = ((const float4*)(fs + (size_t)wave * D_BI))[lane];
  const float4 cw = ((const float4*)conv_w)[lane];
  float x0 = xv.x * cw.x, x1 = xv.y * cw.y, x2 = xv.z * cw.z, x3 = xv.w * cw.w;
  float s  = x0 + x1 + x2 + x3;
  float s2 = x0*x0 + x1*x1 + x2*x2 + x3*x3;
  for (int off = 16; off; off >>= 1) {
    s  += __shfl_xor(s,  off, 32);
    s2 += __shfl_xor(s2, off, 32);
  }
  float mu  = s * (1.0f / D_BI);
  float var = s2 * (1.0f / D_BI) - mu * mu;
  float rs  = rsqrtf(var + EPS_LN);
  const float4 g4 = ((const float4*)ln_g)[lane];
  const float4 b4 = ((const float4*)ln_b)[lane];
  const float4 w4 = ((const float4*)wp_w)[lane];
  float d = ((x0 - mu) * rs * g4.x + b4.x) * w4.x
          + ((x1 - mu) * rs * g4.y + b4.y) * w4.y
          + ((x2 - mu) * rs * g4.z + b4.z) * w4.z
          + ((x3 - mu) * rs * g4.w + b4.w) * w4.w;
  for (int off = 16; off; off >>= 1) d += __shfl_xor(d, off, 32);
  if (lane == 0) {
    float z = d + wp_b[0];
    alpha_out[wave] = SCALE_F / (1.0f + expf(-z));
  }
}

// ---------------------------------------------------------------------------
// 2) Per-batch stats + qty loss.
// ---------------------------------------------------------------------------
__global__ void k_stats(const float* __restrict__ alpha,
                        const int* __restrict__ tlen,
                        float* __restrict__ scale, float* __restrict__ thr,
                        float* __restrict__ absd) {
  int b = blockIdx.x;
  __shared__ float sm[256];
  float s = 0.f;
  for (int t = threadIdx.x; t < T_BI; t += 256) s += alpha[(size_t)b * T_BI + t];
  sm[threadIdx.x] = s; __syncthreads();
  for (int off = 128; off; off >>= 1) {
    if (threadIdx.x < off) sm[threadIdx.x] += sm[threadIdx.x + off];
    __syncthreads();
  }
  if (threadIdx.x == 0) {
    float sum_a = sm[0];
    float tgt   = (float)tlen[b];
    float sc    = tgt / fmaxf(sum_a, 1e-8f);
    float sum_cif = sc * sum_a;
    float cs    = fmaxf(ceilf(sum_cif), 1.0f);
    scale[b] = sc;
    thr[b]   = sum_cif / cs;
    absd[b]  = fabsf(sum_a - tgt);
  }
}

__global__ void k_qty(const float* __restrict__ absd, float* __restrict__ out) {
  __shared__ float sm[64];
  sm[threadIdx.x] = absd[threadIdx.x]; __syncthreads();
  for (int off = 32; off; off >>= 1) {
    if (threadIdx.x < off) sm[threadIdx.x] += sm[threadIdx.x + off];
    __syncthreads();
  }
  if (threadIdx.x == 0) out[0] = sm[0] * (1.0f / B_);
}

// combined bias for the fused output GEMM
__global__ void k_bias3(const float* __restrict__ a, const float* __restrict__ b,
                        const float* __restrict__ c, float* __restrict__ o) {
  int i = threadIdx.x + blockIdx.x * blockDim.x;
  if (i < D_MODEL) o[i] = a[i] + b[i] + c[i];
}

// ---------------------------------------------------------------------------
// 3) Block-per-batch inclusive scan of alpha_cif.
// ---------------------------------------------------------------------------
__global__ void k_cumsum(const float* __restrict__ alpha,
                         const float* __restrict__ scale,
                         float* __restrict__ cum) {
  int b = blockIdx.x, tid = threadIdx.x;
  const float* a = alpha + (size_t)b * T_BI;
  float sc = scale[b];
  __shared__ float part[128];
  const int CH = 24;                       // 128*24 = 3072 >= 3000
  int t0 = tid * CH;
  float s = 0.f;
  for (int i = 0; i < CH; ++i) { int t = t0 + i; if (t < T_BI) s += a[t] * sc; }
  part[tid] = s; __syncthreads();
  for (int off = 1; off < 128; off <<= 1) {
    float v = (tid >= off) ? part[tid - off] : 0.f;
    __syncthreads();
    part[tid] += v;
    __syncthreads();
  }
  float run = part[tid] - s;               // exclusive prefix
  for (int i = 0; i < CH; ++i) {
    int t = t0 + i;
    if (t < T_BI) { run += a[t] * sc; cum[(size_t)b * T_BI + t] = run; }
  }
}

// ---------------------------------------------------------------------------
// 4) CIF integration (gather form, deterministic). Writes bf16 temporal into
//    the K-concatenated activation buffer (row stride K_CAT, col offset 0),
//    and the continuous fire time t_cont.
// ---------------------------------------------------------------------------
__global__ void k_cif_gather(const float* __restrict__ h,
                             const float* __restrict__ alpha,
                             const float* __restrict__ cum,
                             const float* __restrict__ scale,
                             const float* __restrict__ thr,
                             unsigned short* __restrict__ acat,
                             float* __restrict__ t_cont) {
  int r = blockIdx.x;                      // 0 .. M_ROWS-1
  int b = r / N_FIRES, n = r % N_FIRES;
  int ch = threadIdx.x;
  const float* cb = cum + (size_t)b * T_BI;
  float th = thr[b], inv_th = 1.0f / th, sc = scale[b];

  int t0 = lower_bound(cb, T_BI, (float)n * th) - 1;
  if (t0 < 0) t0 = 0;
  int t1 = (n == N_FIRES - 1) ? (T_BI - 1)
                              : min(T_BI - 1, lower_bound(cb, T_BI, (float)(n + 1) * th) + 1);

  float acc = 0.f;
  for (int t = t0; t <= t1; ++t) {
    float c_t = cb[t];
    float p_t = t ? cb[t - 1] : 0.f;
    float cn = c_t * inv_th, pn = p_t * inv_th;
    float kp = floorf(pn), kc = floorf(cn);
    bool fired = kc > kp;
    int ip = min(max((int)kp, 0), N_FIRES - 1);
    float a_t = alpha[(size_t)b * T_BI + t] * sc;
    float w_lo = fired ? (kp + 1.0f - pn) * th : a_t;
    float hv = h[((size_t)b * T_BI + t) * D_BI + ch];
    if (ip == n) acc += w_lo * hv;
    if (fired) {
      int ic = min(max((int)kc, 0), N_FIRES - 1);
      if (ic == n) acc += (cn - kc) * th * hv;
    }
  }
  acat[(size_t)r * K_CAT + ch] = f2bf(acc);

  if (ch == 0) {
    float v = (float)(n + 1) * th;
    int ff = min(lower_bound(cb, T_BI, v), T_BI - 1);
    int tl = max(ff - 1, 0);
    float a_at = alpha[(size_t)b * T_BI + ff] * sc;
    float tc = (float)tl + (v - cb[tl]) / fmaxf(a_at, 1e-8f);
    t_cont[r] = fminf(fmaxf(tc, 0.f), (float)(T_BI - 1));
  }
}

// ---------------------------------------------------------------------------
// 5) Weight pre-swizzle: fp32 [Ksrc x N] slab -> bf16 B-fragment layout at
//    k-block offset kb0 inside a packed buffer with KBtot k-blocks per n-tile.
//    Fragment layout: lane<16 -> col=tn*16+lane, K kb*32+{0..15};
//                     lane>=16 -> col=tn*16+lane-16, K kb*32+{16..31}.
// ---------------------------------------------------------------------------
__global__ void k_pack_w(const float* __restrict__ W, unsigned short* __restrict__ Wp,
                         int Ksrc, int N, int KBtot, int kb0) {
  int idx = blockIdx.x * blockDim.x + threadIdx.x;
  int KBs = Ksrc >> 5, NT = N >> 4;
  if (idx >= NT * KBs * 32) return;
  int lane = idx & 31, region = idx >> 5;
  int kbl = region % KBs, tn = region / KBs;
  int col = tn * 16 + (lane & 15);
  int kbase = kbl * 32 + ((lane >= 16) ? 16 : 0);
  unsigned short* dst = Wp + ((size_t)tn * KBtot + kb0 + kbl) * 512 + lane * 16;
  for (int j = 0; j < 16; ++j) dst[j] = f2bf(W[(size_t)(kbase + j) * N + col]);
}

// ---------------------------------------------------------------------------
// 6) WMMA GEMM: C[M,N] = A[M,K](bf16, row stride lda) @ W(bf16 packed) + bias.
//    One wave computes 16 rows x 128 cols (8 n-tiles): each A fragment is
//    reused by 8 v_wmma_f32_16x16x32_bf16 issues.
// ---------------------------------------------------------------------------
__global__ void k_gemm_bf16(const unsigned short* __restrict__ A,
                            const unsigned short* __restrict__ Wp,
                            const float* __restrict__ bias,
                            float* __restrict__ C,
                            int N, int K, int lda) {
  int wave = blockIdx.x * (blockDim.x >> 5) + (threadIdx.x >> 5);
  int lane = threadIdx.x & 31;
  int ngrp = N >> 7;                       // groups of 8 n-tiles (128 cols)
  int tm = wave / ngrp, tg = wave % ngrp;
  int tn0 = tg * 8;
  int KB = K >> 5;

  v8f acc[8];
#pragma unroll
  for (int j = 0; j < 8; ++j) acc[j] = (v8f){0.f,0.f,0.f,0.f,0.f,0.f,0.f,0.f};

  int row  = tm * 16 + (lane & 15);
  int aofs = (lane >= 16) ? 16 : 0;
  const unsigned short* pA = A + (size_t)row * lda + aofs;
  const unsigned short* pB = Wp + ((size_t)tn0 * KB) * 512 + lane * 16;

  for (int kb = 0; kb < KB; ++kb) {
    union { v16bf v; uint4 q[2]; } af;
    const unsigned short* pa = pA + kb * 32;
    af.q[0] = *(const uint4*)(pa);
    af.q[1] = *(const uint4*)(pa + 8);
#pragma unroll
    for (int j = 0; j < 8; ++j) {
      union { v16bf v; uint4 q[2]; } bf;
      const unsigned short* pb = pB + ((size_t)j * KB + kb) * 512;
      bf.q[0] = *(const uint4*)(pb);
      bf.q[1] = *(const uint4*)(pb + 8);
      acc[j] = __builtin_amdgcn_wmma_f32_16x16x32_bf16(
          false, af.v, false, bf.v, (short)0, acc[j], false, false);
    }
  }

  int hi = lane >> 4, lo = lane & 15;
#pragma unroll
  for (int j = 0; j < 8; ++j) {
    int col = (tn0 + j) * 16 + lo;
    float bs = bias[col];
#pragma unroll
    for (int v = 0; v < 8; ++v) {
      int r = tm * 16 + v + 8 * hi;
      C[(size_t)r * N + col] = acc[j][v] + bs;
    }
  }
}

// ---------------------------------------------------------------------------
// 7) Interp + FiLM -> bf16 column slab of the concatenated activation buffer.
// ---------------------------------------------------------------------------
__global__ void k_film_interp(const float* __restrict__ src, int Ts,
                              const float* __restrict__ film,
                              const float* __restrict__ tc,
                              unsigned short* __restrict__ acat, int colofs) {
  int r = blockIdx.x, d = threadIdx.x;     // D_SW threads
  int b = r / N_FIRES;
  float t = tc[r] * ((float)Ts / (float)T_BI);
  int lo = min(max((int)t, 0), Ts - 2);
  float w = t - (float)lo;
  const float* s = src + ((size_t)b * Ts + lo) * D_SW + d;
  float val = (1.0f - w) * s[0] + w * s[D_SW];
  float g  = film[(size_t)r * (2 * D_SW) + d];
  float bb = film[(size_t)r * (2 * D_SW) + D_SW + d];
  acat[(size_t)r * K_CAT + colofs + d] = f2bf(g * val + bb);
}

// ---------------------------------------------------------------------------
// Launch
// ---------------------------------------------------------------------------
extern "C" void kernel_launch(void* const* d_in, const int* in_sizes, int n_in,
                              void* d_out, int out_size, void* d_ws, size_t ws_size,
                              hipStream_t stream) {
  (void)in_sizes; (void)n_in; (void)out_size; (void)ws_size;
  const float* fire   = (const float*)d_in[0];
  const float* src0   = (const float*)d_in[1];
  const float* src1   = (const float*)d_in[2];
  const int*   tlen   = (const int*)  d_in[3];
  const float* conv_w = (const float*)d_in[4];
  const float* ln_g   = (const float*)d_in[5];
  const float* ln_b   = (const float*)d_in[6];
  const float* wp_w   = (const float*)d_in[7];
  const float* wp_b   = (const float*)d_in[8];
  const float* f0w    = (const float*)d_in[9];
  const float* f0b    = (const float*)d_in[10];
  const float* f1w    = (const float*)d_in[11];
  const float* f1b    = (const float*)d_in[12];
  const float* tpw    = (const float*)d_in[13];
  const float* tpb    = (const float*)d_in[14];
  const float* apw    = (const float*)d_in[15];
  const float* apb    = (const float*)d_in[16];
  const float* bpw    = (const float*)d_in[17];
  const float* bpb    = (const float*)d_in[18];

  float* embs  = (float*)d_out;                    // [M_ROWS, 512]
  float* alpha = embs + EMB_SZ;                    // [B, T_BI]
  float* qty   = alpha + ALPHA_SZ;                 // [1]

  // workspace carve-up (all chunk sizes multiples of 256 bytes)
  char* w = (char*)d_ws;
  float*          cum   = (float*)w;          w += (size_t)ALPHA_SZ * 4;
  float*          scale = (float*)w;          w += 256;
  float*          thr   = (float*)w;          w += 256;
  float*          absd  = (float*)w;          w += 256;
  float*          cbias = (float*)w;          w += 2048;
  float*          tcont = (float*)w;          w += (size_t)M_ROWS * 4;
  unsigned short* acat  = (unsigned short*)w; w += (size_t)M_ROWS * K_CAT * 2;   // 19.7MB
  float*          film  = (float*)w;          w += (size_t)M_ROWS * (2 * D_SW) * 4; // 29.5MB
  unsigned short* f0p   = (unsigned short*)w; w += (size_t)4 * 24 * 512 * 2;
  unsigned short* f1p   = (unsigned short*)w; w += (size_t)4 * 24 * 512 * 2;
  unsigned short* wcp   = (unsigned short*)w; // 16 * 32 * 512 bf16 = 512KB

  // weight pre-swizzles
  k_pack_w<<<12, 256, 0, stream>>>(f0w, f0p, D_BI, 2 * D_SW, 4, 0);
  k_pack_w<<<12, 256, 0, stream>>>(f1w, f1p, D_BI, 2 * D_SW, 4, 0);
  k_pack_w<<<16, 256, 0, stream>>>(tpw, wcp, D_BI, D_MODEL, 16, 0);   // K rows   0..127
  k_pack_w<<<24, 256, 0, stream>>>(apw, wcp, D_SW, D_MODEL, 16, 4);   // K rows 128..319
  k_pack_w<<<24, 256, 0, stream>>>(bpw, wcp, D_SW, D_MODEL, 16, 10);  // K rows 320..511
  k_bias3<<<2, 256, 0, stream>>>(tpb, apb, bpb, cbias);

  // alpha predictor + stats + scan
  k_alpha<<<(B_ * T_BI) / 8, 256, 0, stream>>>(fire, conv_w, ln_g, ln_b, wp_w, wp_b, alpha);
  k_stats<<<B_, 256, 0, stream>>>(alpha, tlen, scale, thr, absd);
  k_qty<<<1, 64, 0, stream>>>(absd, qty);
  k_cumsum<<<B_, 128, 0, stream>>>(alpha, scale, cum);

  // CIF gather + fire times (fills acat cols 0..127)
  k_cif_gather<<<M_ROWS, D_BI, 0, stream>>>(fire, alpha, cum, scale, thr, acat, tcont);

  // film0 -> pitch slab (cols 128..319); film1 -> beat slab (cols 320..511)
  // film GEMMs: M/16 * (384/128) = 1200*3 = 3600 waves -> 450 blocks
  k_gemm_bf16<<<450, 256, 0, stream>>>(acat, f0p, f0b, film, 2 * D_SW, D_BI, K_CAT);
  k_film_interp<<<M_ROWS, D_SW, 0, stream>>>(src0, T_SW, film, tcont, acat, D_BI);
  k_gemm_bf16<<<450, 256, 0, stream>>>(acat, f1p, f1b, film, 2 * D_SW, D_BI, K_CAT);
  k_film_interp<<<M_ROWS, D_SW, 0, stream>>>(src1, T_S1, film, tcont, acat, D_BI + D_SW);

  // fused output projection: [19200 x 512] @ [512 x 512] + cbias
  // 1200 * (512/128) = 4800 waves -> 600 blocks
  k_gemm_bf16<<<600, 256, 0, stream>>>(acat, wcp, cbias, embs, D_MODEL, K_CAT, K_CAT);
}
